// QSAR_7284264534171
// MI455X (gfx1250) — matrix operand order repaired and checked by reading, not compiled, not run on hardware
//
#include <hip/hip_runtime.h>
#include <hip/hip_bf16.h>

// ---------- types ----------
typedef __attribute__((ext_vector_type(16))) __bf16 v16bf;
typedef __attribute__((ext_vector_type(8)))  __bf16 v8bf;
typedef __attribute__((ext_vector_type(8)))  float  v8f;

// ---------- problem constants ----------
#define HDIM   384
#define NTILES 24          // HDIM/16
#define NB     131073      // 1 + M*4*A bonds
#define NBP    131088      // 8193*16 padded
#define NA     32769       // 1 + M*A atoms
#define NAP    32784       // 2049*16 padded
#define NMOL   1024
#define NATOM  32
#define KT_I   5           // ceil(150/32)
#define KT_H   12          // 384/32
#define KT_O   17          // ceil(523/32)

// ---------- helpers ----------
__device__ inline v8f vzero8() {
  v8f z = {0.f,0.f,0.f,0.f,0.f,0.f,0.f,0.f};
  return z;
}

__device__ inline v16bf mk16(v8bf lo, v8bf hi) {
  v16bf r;
#pragma unroll
  for (int i = 0; i < 8; ++i) { r[i] = lo[i]; r[i + 8] = hi[i]; }
  return r;
}

__device__ inline v8f wmma_bf16(v16bf a, v16bf b, v8f c) {
  // D = A(16x32 bf16) * B(32x16 bf16) + C(16x16 f32)
  return __builtin_amdgcn_wmma_f32_16x16x32_bf16(
      false, a, false, b, (short)0, c, false, false);
}

// =====================================================================
// Pack a KxN (N=384) row-major fp32 weight into per-lane WMMA B-fragment
// layout, bf16, zero-padded in K:
//   dst[((nt*KT + kt)*32 + lane)*16 + e] = W[(kt*32 + (lane>>4)*8 + (e&7) + (e>>3)*16)*384
//                                            + nt*16 + (lane&15)]
// =====================================================================
__global__ void pack_weight_kernel(const float* __restrict__ W, int K, int KT,
                                   __bf16* __restrict__ dst) {
  int id = blockIdx.x * 256 + threadIdx.x;
  int total = NTILES * KT * 512;
  if (id >= total) return;
  int blk = id >> 9;
  int within = id & 511;
  int lane = within >> 4;
  int e = within & 15;
  int nt = blk / KT;
  int kt = blk - nt * KT;
  int col = nt * 16 + (lane & 15);
  int k = kt * 32 + (lane >> 4) * 8 + (e & 7) + (e >> 3) * 16;
  float v = (k < K) ? W[(size_t)k * HDIM + col] : 0.0f;
  dst[(size_t)blk * 512 + within] = (__bf16)v;
}

// =====================================================================
// inp = f_bonds @ W_i ; message0 = relu(inp).  Both stored bf16.
// grid (8193, 3), block 256 (8 waves, one 16x16 n-tile per wave)
// =====================================================================
__global__ void bond_input_kernel(const float* __restrict__ fb,
                                  const __bf16* __restrict__ pWi,
                                  __bf16* __restrict__ inp_bf,
                                  __bf16* __restrict__ msg) {
  int lane = threadIdx.x & 31;
  int wave = threadIdx.x >> 5;
  int nt = blockIdx.y * 8 + wave;
  int row0 = blockIdx.x * 16;
  int m15 = lane & 15;
  int kh = lane >> 4;
  int r = row0 + m15;
  int rs = (r < NB) ? r : 0;
  const float* arow = fb + (size_t)rs * 150;

  v8f acc = vzero8();
#pragma unroll
  for (int kt = 0; kt < KT_I; ++kt) {
    int k0 = kt * 32 + kh * 8;
    v16bf a;
#pragma unroll
    for (int e = 0; e < 8; ++e) {
      int ka = k0 + e;
      int kb = k0 + 16 + e;
      float va = (ka < 150) ? arow[ka] : 0.f;
      float vb = (kb < 150) ? arow[kb] : 0.f;
      a[e] = (__bf16)va;
      a[e + 8] = (__bf16)vb;
    }
    v16bf b = ((const v16bf*)pWi)[(size_t)(nt * KT_I + kt) * 32 + lane];
    acc = wmma_bf16(a, b, acc);
  }

  int col = nt * 16 + m15;
  int mb = kh * 8;
#pragma unroll
  for (int i = 0; i < 8; ++i) {
    int rr = row0 + mb + i;
    if (rr < NB) {
      float v = acc[i];
      inp_bf[(size_t)rr * HDIM + col] = (__bf16)v;
      msg[(size_t)rr * HDIM + col] = (__bf16)fmaxf(v, 0.f);
    }
  }
}

// =====================================================================
// a_msg[a][h] = sum_{j<4} message[a2b[a][j]][h]   (fp32 out)
// grid NAP blocks x 384 threads
// =====================================================================
__global__ void amsg_kernel(const __bf16* __restrict__ msg,
                            const int* __restrict__ a2b,
                            float* __restrict__ amsg) {
  int atom = blockIdx.x;
  int h = threadIdx.x;
  float s = 0.f;
  if (atom < NA) {
    const int* nb = a2b + (size_t)atom * 4;
#pragma unroll
    for (int j = 0; j < 4; ++j) {
      int b = nb[j];
      s += (float)msg[(size_t)b * HDIM + h];
    }
  }
  amsg[(size_t)atom * HDIM + h] = s;
}

// =====================================================================
// message_new = relu(inp + (a_msg[b2a] - message[b2revb]) @ W_h[d])
// Gather is fused into the A-fragment load.  grid (8193,3), block 256.
// =====================================================================
__global__ void msg_step_kernel(const __bf16* __restrict__ inp_bf,
                                const __bf16* __restrict__ msgIn,
                                const float* __restrict__ amsg,
                                const int* __restrict__ b2a,
                                const int* __restrict__ b2revb,
                                const __bf16* __restrict__ pWh,
                                __bf16* __restrict__ msgOut) {
  int lane = threadIdx.x & 31;
  int wave = threadIdx.x >> 5;
  int nt = blockIdx.y * 8 + wave;
  int row0 = blockIdx.x * 16;
  int m15 = lane & 15;
  int kh = lane >> 4;
  int r = row0 + m15;
  int rs = (r < NB) ? r : 0;
  int ba = b2a[rs];
  int brev = b2revb[rs];
  const float* pa = amsg + (size_t)ba * HDIM;
  const __bf16* pm = msgIn + (size_t)brev * HDIM;

  v8f acc = vzero8();
#pragma unroll
  for (int kt = 0; kt < KT_H; ++kt) {
    int k0 = kt * 32 + kh * 8;
    v8f a0 = *(const v8f*)(pa + k0);
    v8f a1 = *(const v8f*)(pa + k0 + 16);
    v8bf q0 = *(const v8bf*)(pm + k0);
    v8bf q1 = *(const v8bf*)(pm + k0 + 16);
    v16bf a;
#pragma unroll
    for (int e = 0; e < 8; ++e) {
      a[e] = (__bf16)(a0[e] - (float)q0[e]);
      a[e + 8] = (__bf16)(a1[e] - (float)q1[e]);
    }
    v16bf b = ((const v16bf*)pWh)[(size_t)(nt * KT_H + kt) * 32 + lane];
    acc = wmma_bf16(a, b, acc);
  }

  int col = nt * 16 + m15;
  int mb = kh * 8;
#pragma unroll
  for (int i = 0; i < 8; ++i) {
    int rr = row0 + mb + i;
    if (rr < NB) {
      float v = acc[i] + (float)inp_bf[(size_t)rr * HDIM + col];
      msgOut[(size_t)rr * HDIM + col] = (__bf16)fmaxf(v, 0.f);
    }
  }
}

// =====================================================================
// atom_h = relu(concat(f_atoms, a_msg) @ W_o + b_o)
// cur[m*32+a] = atom_h[1 + m*32 + a]   (fp32 + bf16 copies)
// grid (2049, 3), block 256.  K = 523 padded to 544.
// =====================================================================
__global__ void atom_out_kernel(const float* __restrict__ f_atoms,
                                const float* __restrict__ amsg,
                                const float* __restrict__ b_o,
                                const __bf16* __restrict__ pWo,
                                float* __restrict__ curF,
                                __bf16* __restrict__ curB) {
  int lane = threadIdx.x & 31;
  int wave = threadIdx.x >> 5;
  int nt = blockIdx.y * 8 + wave;
  int row0 = blockIdx.x * 16;
  int m15 = lane & 15;
  int kh = lane >> 4;
  int r = row0 + m15;
  int rs = (r < NA) ? r : 0;
  const float* fa = f_atoms + (size_t)rs * 139;
  const float* am = amsg + (size_t)rs * HDIM;

  v8f acc = vzero8();
#pragma unroll
  for (int kt = 0; kt < KT_O; ++kt) {
    int k0 = kt * 32 + kh * 8;
    v16bf a;
#pragma unroll
    for (int e = 0; e < 8; ++e) {
      int ka = k0 + e;
      int kb = k0 + 16 + e;
      float va = (ka < 139) ? fa[ka] : ((ka < 523) ? am[ka - 139] : 0.f);
      float vb = (kb < 139) ? fa[kb] : ((kb < 523) ? am[kb - 139] : 0.f);
      a[e] = (__bf16)va;
      a[e + 8] = (__bf16)vb;
    }
    v16bf b = ((const v16bf*)pWo)[(size_t)(nt * KT_O + kt) * 32 + lane];
    acc = wmma_bf16(a, b, acc);
  }

  int col = nt * 16 + m15;
  float bo = b_o[col];
  int mb = kh * 8;
#pragma unroll
  for (int i = 0; i < 8; ++i) {
    int rr = row0 + mb + i;
    if (rr >= 1 && rr < NA) {
      float v = fmaxf(acc[i] + bo, 0.f);
      size_t idx = (size_t)(rr - 1) * HDIM + col;
      curF[idx] = v;
      curB[idx] = (__bf16)v;
    }
  }
}

// =====================================================================
// Per-molecule fused attention readout. One workgroup (256 threads,
// 8 waves) per molecule.
//   Q = cur @ W_a ; S = Q @ cur^T ; att = softmax(S)
//   P = att @ cur ; att_h = relu(P @ W_b + b_b)
//   out[m] = sum_a (cur + att_h)
// =====================================================================
__global__ void attn_kernel(const __bf16* __restrict__ curB,
                            const float* __restrict__ curF,
                            const __bf16* __restrict__ pWa,
                            const __bf16* __restrict__ pWb,
                            const float* __restrict__ b_b,
                            float* __restrict__ out) {
  __shared__ __align__(32) __bf16 sCurT[HDIM * NATOM];   // cur^T  (h-major)
  __shared__ __align__(32) __bf16 sQ[NATOM * HDIM];      // Q, then reused as P
  __shared__ __align__(32) float  sS[NATOM * NATOM];
  __shared__ __align__(32) __bf16 sAtt[NATOM * NATOM];
  __shared__ __align__(32) float  sOut[HDIM];

  int mol = blockIdx.x;
  const __bf16* cb = curB + (size_t)mol * NATOM * HDIM;
  const float* cf = curF + (size_t)mol * NATOM * HDIM;
  int tid = threadIdx.x;
  int lane = tid & 31;
  int wave = tid >> 5;
  int m15 = lane & 15;
  int kh = lane >> 4;
  int mb = kh * 8;

  // stage 1: build cur^T in LDS, zero output accumulator
  for (int i = tid; i < NATOM * HDIM; i += 256) {
    int a = i / HDIM;
    int h = i - a * HDIM;
    sCurT[h * NATOM + a] = cb[i];
  }
  for (int i = tid; i < HDIM; i += 256) sOut[i] = 0.f;
  __syncthreads();

  // stage 2: Q = cur @ W_a  (2 x 24 tiles over 8 waves)
#pragma unroll
  for (int j = 0; j < 6; ++j) {
    int t = wave * 6 + j;
    int mt = t / NTILES;
    int nt = t - mt * NTILES;
    const __bf16* arow = cb + (size_t)(mt * 16 + m15) * HDIM;
    v8f acc = vzero8();
#pragma unroll
    for (int kt = 0; kt < KT_H; ++kt) {
      int k0 = kt * 32 + kh * 8;
      v16bf a = mk16(*(const v8bf*)(arow + k0), *(const v8bf*)(arow + k0 + 16));
      v16bf b = ((const v16bf*)pWa)[(size_t)(nt * KT_H + kt) * 32 + lane];
      acc = wmma_bf16(a, b, acc);
    }
    int col = nt * 16 + m15;
#pragma unroll
    for (int i = 0; i < 8; ++i)
      sQ[(mt * 16 + mb + i) * HDIM + col] = (__bf16)acc[i];
  }
  __syncthreads();

  // stage 3: S = Q @ cur^T  (32x32 over waves 0..3)
  if (wave < 4) {
    int mt = wave >> 1;
    int nt = wave & 1;
    const __bf16* arow = sQ + (size_t)(mt * 16 + m15) * HDIM;
    const __bf16* brow = cb + (size_t)(nt * 16 + m15) * HDIM;  // B[k][n]=cur[n][k]
    v8f acc = vzero8();
#pragma unroll
    for (int kt = 0; kt < KT_H; ++kt) {
      int ka = kt * 32 + kh * 8;
      v16bf a = mk16(*(const v8bf*)(arow + ka), *(const v8bf*)(arow + ka + 16));
      v16bf b = mk16(*(const v8bf*)(brow + ka), *(const v8bf*)(brow + ka + 16));
      acc = wmma_bf16(a, b, acc);
    }
#pragma unroll
    for (int i = 0; i < 8; ++i)
      sS[(mt * 16 + mb + i) * NATOM + nt * 16 + m15] = acc[i];
  }
  __syncthreads();

  // stage 4: row softmax (32 rows, one thread each)
  if (tid < NATOM) {
    float mx = -3.4e38f;
    for (int b = 0; b < NATOM; ++b) mx = fmaxf(mx, sS[tid * NATOM + b]);
    float sum = 0.f;
    for (int b = 0; b < NATOM; ++b) {
      float ev = expf(sS[tid * NATOM + b] - mx);
      sS[tid * NATOM + b] = ev;
      sum += ev;
    }
    float inv = 1.f / sum;
    for (int b = 0; b < NATOM; ++b)
      sAtt[tid * NATOM + b] = (__bf16)(sS[tid * NATOM + b] * inv);
  }
  __syncthreads();

  // stage 5: P = att @ cur  (K = 32, single WMMA step) -> overwrite sQ
#pragma unroll
  for (int j = 0; j < 6; ++j) {
    int t = wave * 6 + j;
    int mt = t / NTILES;
    int nt = t - mt * NTILES;
    const __bf16* arow = sAtt + (mt * 16 + m15) * NATOM;
    v16bf a = mk16(*(const v8bf*)(arow + kh * 8),
                   *(const v8bf*)(arow + kh * 8 + 16));
    v16bf b = mk16(*(const v8bf*)(sCurT + (nt * 16 + m15) * NATOM + kh * 8),
                   *(const v8bf*)(sCurT + (nt * 16 + m15) * NATOM + kh * 8 + 16));
    v8f acc = vzero8();
    acc = wmma_bf16(a, b, acc);
    int col = nt * 16 + m15;
#pragma unroll
    for (int i = 0; i < 8; ++i)
      sQ[(mt * 16 + mb + i) * HDIM + col] = (__bf16)acc[i];
  }
  __syncthreads();

  // stage 6: Y = P @ W_b ; relu(+b_b) ; add cur ; reduce over atoms
#pragma unroll
  for (int j = 0; j < 6; ++j) {
    int t = wave * 6 + j;
    int mt = t / NTILES;
    int nt = t - mt * NTILES;
    const __bf16* arow = sQ + (size_t)(mt * 16 + m15) * HDIM;
    v8f acc = vzero8();
#pragma unroll
    for (int kt = 0; kt < KT_H; ++kt) {
      int k0 = kt * 32 + kh * 8;
      v16bf a = mk16(*(const v8bf*)(arow + k0), *(const v8bf*)(arow + k0 + 16));
      v16bf b = ((const v16bf*)pWb)[(size_t)(nt * KT_H + kt) * 32 + lane];
      acc = wmma_bf16(a, b, acc);
    }
    int col = nt * 16 + m15;
    float bb = b_b[col];
    float partial = 0.f;
#pragma unroll
    for (int i = 0; i < 8; ++i) {
      int a_ = mt * 16 + mb + i;
      float y = fmaxf(acc[i] + bb, 0.f);
      partial += y + cf[(size_t)a_ * HDIM + col];
    }
    atomicAdd(&sOut[col], partial);
  }
  __syncthreads();

  for (int i = tid; i < HDIM; i += 256)
    out[(size_t)mol * HDIM + i] = sOut[i];
}

// =====================================================================
// host launcher
// =====================================================================
static constexpr size_t SZ_PWI  = (size_t)NTILES * KT_I * 1024;
static constexpr size_t SZ_PWH1 = (size_t)NTILES * KT_H * 1024;
static constexpr size_t SZ_PWH  = 3 * SZ_PWH1;
static constexpr size_t SZ_PWO  = (size_t)NTILES * KT_O * 1024;
static constexpr size_t SZ_PWA  = (size_t)NTILES * KT_H * 1024;
static constexpr size_t SZ_MSG  = (size_t)NBP * HDIM * 2;
static constexpr size_t SZ_AMSG = (size_t)NAP * HDIM * 4;
static constexpr size_t SZ_CURF = (size_t)NMOL * NATOM * HDIM * 4;

static constexpr size_t OFF_PWI  = 0;
static constexpr size_t OFF_PWH  = OFF_PWI + SZ_PWI;
static constexpr size_t OFF_PWO  = OFF_PWH + SZ_PWH;
static constexpr size_t OFF_PWA  = OFF_PWO + SZ_PWO;
static constexpr size_t OFF_PWB  = OFF_PWA + SZ_PWA;
static constexpr size_t OFF_INP  = OFF_PWB + SZ_PWA;
static constexpr size_t OFF_MSGA = OFF_INP + SZ_MSG;
static constexpr size_t OFF_MSGB = OFF_MSGA + SZ_MSG;
static constexpr size_t OFF_AMSG = OFF_MSGB + SZ_MSG;
static constexpr size_t OFF_CURF = OFF_AMSG + SZ_AMSG;
static constexpr size_t OFF_CURB = OFF_CURF + SZ_CURF;

extern "C" void kernel_launch(void* const* d_in, const int* in_sizes, int n_in,
                              void* d_out, int out_size, void* d_ws, size_t ws_size,
                              hipStream_t stream) {
  (void)in_sizes; (void)n_in; (void)out_size; (void)ws_size;
  const float* f_atoms = (const float*)d_in[0];
  const float* f_bonds = (const float*)d_in[1];
  const float* W_i     = (const float*)d_in[2];
  const float* W_h     = (const float*)d_in[3];   // (3, 384, 384)
  const float* W_o     = (const float*)d_in[4];   // (523, 384)
  const float* b_o     = (const float*)d_in[5];
  const float* W_a     = (const float*)d_in[6];
  const float* W_b     = (const float*)d_in[7];
  const float* b_b     = (const float*)d_in[8];
  const int*   a2b     = (const int*)d_in[9];
  const int*   b2a     = (const int*)d_in[10];
  const int*   b2revb  = (const int*)d_in[11];
  float* out = (float*)d_out;

  char* ws = (char*)d_ws;
  __bf16* pWi  = (__bf16*)(ws + OFF_PWI);
  __bf16* pWh  = (__bf16*)(ws + OFF_PWH);
  __bf16* pWo  = (__bf16*)(ws + OFF_PWO);
  __bf16* pWa  = (__bf16*)(ws + OFF_PWA);
  __bf16* pWb  = (__bf16*)(ws + OFF_PWB);
  __bf16* inp  = (__bf16*)(ws + OFF_INP);
  __bf16* msgA = (__bf16*)(ws + OFF_MSGA);
  __bf16* msgB = (__bf16*)(ws + OFF_MSGB);
  float*  amsg = (float*)(ws + OFF_AMSG);
  float*  curF = (float*)(ws + OFF_CURF);
  __bf16* curB = (__bf16*)(ws + OFF_CURB);

  // pack weights into WMMA B-fragment layout (bf16)
  pack_weight_kernel<<<(NTILES * KT_I * 512 + 255) / 256, 256, 0, stream>>>(
      W_i, 150, KT_I, pWi);
  for (int d = 0; d < 3; ++d)
    pack_weight_kernel<<<(NTILES * KT_H * 512 + 255) / 256, 256, 0, stream>>>(
        W_h + (size_t)d * HDIM * HDIM, HDIM, KT_H,
        pWh + (size_t)d * NTILES * KT_H * 512);
  pack_weight_kernel<<<(NTILES * KT_O * 512 + 255) / 256, 256, 0, stream>>>(
      W_o, 523, KT_O, pWo);
  pack_weight_kernel<<<(NTILES * KT_H * 512 + 255) / 256, 256, 0, stream>>>(
      W_a, HDIM, KT_H, pWa);
  pack_weight_kernel<<<(NTILES * KT_H * 512 + 255) / 256, 256, 0, stream>>>(
      W_b, HDIM, KT_H, pWb);

  dim3 gbond(NBP / 16, 3);
  bond_input_kernel<<<gbond, 256, 0, stream>>>(f_bonds, pWi, inp, msgA);

  __bf16* cur = msgA;
  __bf16* nxt = msgB;
  for (int d = 0; d < 3; ++d) {
    amsg_kernel<<<NAP, HDIM, 0, stream>>>(cur, a2b, amsg);
    msg_step_kernel<<<gbond, 256, 0, stream>>>(
        inp, cur, amsg, b2a, b2revb,
        pWh + (size_t)d * NTILES * KT_H * 512, nxt);
    __bf16* t = cur; cur = nxt; nxt = t;
  }
  amsg_kernel<<<NAP, HDIM, 0, stream>>>(cur, a2b, amsg);

  dim3 gatom(NAP / 16, 3);
  atom_out_kernel<<<gatom, 256, 0, stream>>>(f_atoms, amsg, b_o, pWo, curF, curB);

  attn_kernel<<<NMOL, 256, 0, stream>>>(curB, curF, pWa, pWb, b_b, out);
}
